// BiMambaBlock_35871566856361
// MI455X (gfx1250) — compile-verified, matrix operand order checked by reading, not run on hardware
//
#include <hip/hip_runtime.h>
#include <hip/hip_bf16.h>

// ---------------------------------------------------------------------------
// BiMamba block for MI455X (gfx1250, wave32, WMMA).
// GEMMs: bf16 x bf16 -> f32 via v_wmma_f32_16x16x32_bf16 (direct b128 loads,
//        compiler clauses + partial loadcnt waits pipeline them vs WMMA).
// Scan:  fp32 VALU recurrence, double-buffered GLOBAL_LOAD_ASYNC_TO_LDS
//        staging (ASYNCcnt) so the serial chain never waits on memory.
// ---------------------------------------------------------------------------

#define B_ 4
#define L_ 4096
#define DM_ 512           // D_MODEL
#define DI_ 1024          // D_INNER
#define DS_ 64            // D_STATE
#define NH_ 16            // NHEADS
#define HD_ 64            // HEADDIM
#define CONVD_ 1152       // CONV_DIM = DI + 2*DS
#define NPROJ_ 2192       // D_IN_PROJ
#define NPROJ_PAD_ 2240   // padded to multiple of 64
#define ROWS_ (B_ * L_)   // 16384
#define EPS_ 1e-5f

typedef __attribute__((ext_vector_type(16))) __bf16 v16bf;
typedef __attribute__((ext_vector_type(8)))  __bf16 v8bf;
typedef __attribute__((ext_vector_type(8)))  float  v8f;

static __device__ __forceinline__ float siluf(float x) {
  return x / (1.0f + __expf(-x));
}

// ---------------------------------------------------------------------------
// bf16 conversions / weight prep
// ---------------------------------------------------------------------------
__global__ void cvt_f32_bf16_kernel(const float* __restrict__ s,
                                    __bf16* __restrict__ d, int n) {
  int i = blockIdx.x * blockDim.x + threadIdx.x;
  if (i < n) d[i] = (__bf16)s[i];
}

// in_proj_w (2192 x 512) -> bf16 padded to (2240 x 512), zero pad rows
__global__ void pad_inproj_kernel(const float* __restrict__ s,
                                  __bf16* __restrict__ d) {
  int i = blockIdx.x * blockDim.x + threadIdx.x;
  if (i >= NPROJ_PAD_ * DM_) return;
  int n = i / DM_, k = i % DM_;
  d[i] = (n < NPROJ_) ? (__bf16)s[n * DM_ + k] : (__bf16)0.0f;
}

// cat([y,y]) @ w1^T  ==  y @ (w1[:, :512] + w1[:, 512:])^T  -> fold once
__global__ void fold_w1_kernel(const float* __restrict__ w1,
                               __bf16* __restrict__ d) {
  int i = blockIdx.x * blockDim.x + threadIdx.x;
  if (i >= 1024 * DM_) return;
  int n = i / DM_, k = i % DM_;
  d[i] = (__bf16)(w1[n * 1024 + k] + w1[n * 1024 + DM_ + k]);
}

// ---------------------------------------------------------------------------
// WMMA fragment loader (ISA 7.12.2, 16-bit A/B 16x32 layout):
// lane<16: elems 0..7 -> K = kk..kk+7 ; elems 8..15 -> K = kk+16..kk+23
// lane>=16 uses kk += 8. Two contiguous 16B chunks per lane.
// ---------------------------------------------------------------------------
static __device__ __forceinline__ v16bf load_frag(const __bf16* __restrict__ row,
                                                  int kk) {
  v8bf lo = *(const v8bf*)(row + kk);
  v8bf hi = *(const v8bf*)(row + kk + 16);
  v16bf f;
#pragma unroll
  for (int i = 0; i < 8; ++i) { f[i] = lo[i]; f[i + 8] = hi[i]; }
  return f;
}

// C = A(MxK, bf16 row-major) * W(NxK, bf16 row-major)^T  (+bias) (+silu)
// block = 256 thr = 8 waves in 4x2; block tile 128(M) x 64(N); wave tile 32x32.
// mode 0: store fp32 to outF ; mode 1: silu then store bf16 to outB.
__global__ void gemm_bf16_wmma_kernel(const __bf16* __restrict__ A,
                                      const __bf16* __restrict__ W,
                                      const float* __restrict__ bias,
                                      float* __restrict__ outF,
                                      __bf16* __restrict__ outB,
                                      int K, int ldc, int mode) {
  const int tid  = threadIdx.x;
  const int lane = tid & 31;
  const int wave = tid >> 5;
  const int wm   = wave >> 1;          // 0..3
  const int wn   = wave & 1;           // 0..1
  const int mBase = blockIdx.x * 128 + wm * 32;
  const int nBase = blockIdx.y * 64  + wn * 32;
  const int mrow = lane & 15;
  const int half = lane >> 4;
  const int koff = half * 8;

  v8f acc[2][2] = {};

  const __bf16* a0p = A + (size_t)(mBase +      mrow) * K;
  const __bf16* a1p = A + (size_t)(mBase + 16 + mrow) * K;
  const __bf16* b0p = W + (size_t)(nBase +      mrow) * K;
  const __bf16* b1p = W + (size_t)(nBase + 16 + mrow) * K;

  for (int k0 = 0; k0 < K; k0 += 32) {
    v16bf a0 = load_frag(a0p, k0 + koff);
    v16bf a1 = load_frag(a1p, k0 + koff);
    v16bf b0 = load_frag(b0p, k0 + koff);
    v16bf b1 = load_frag(b1p, k0 + koff);
    acc[0][0] = __builtin_amdgcn_wmma_f32_16x16x32_bf16(false, a0, false, b0,
                                                        (short)0, acc[0][0], false, false);
    acc[0][1] = __builtin_amdgcn_wmma_f32_16x16x32_bf16(false, a0, false, b1,
                                                        (short)0, acc[0][1], false, false);
    acc[1][0] = __builtin_amdgcn_wmma_f32_16x16x32_bf16(false, a1, false, b0,
                                                        (short)0, acc[1][0], false, false);
    acc[1][1] = __builtin_amdgcn_wmma_f32_16x16x32_bf16(false, a1, false, b1,
                                                        (short)0, acc[1][1], false, false);
  }

  // C/D layout: VGPR v -> row v + 8*half ; col = lane&15 (per 16x16 tile)
#pragma unroll
  for (int ti = 0; ti < 2; ++ti) {
#pragma unroll
    for (int tj = 0; tj < 2; ++tj) {
      const int col = nBase + tj * 16 + (lane & 15);
      const float bv = bias ? bias[col] : 0.0f;
#pragma unroll
      for (int v = 0; v < 8; ++v) {
        const int row = mBase + ti * 16 + v + 8 * half;
        const float val = acc[ti][tj][v] + bv;
        if (mode == 1) {
          outB[(size_t)row * ldc + col] = (__bf16)siluf(val);
        } else {
          outF[(size_t)row * ldc + col] = val;
        }
      }
    }
  }
}

// ---------------------------------------------------------------------------
// Anti-causal depthwise conv (reversal folded in) + SiLU.
// xc[l,c] = silu( conv_b[c] + sum_k conv_w[c,k] * zx[l+3-k, 1024+c] )
// ---------------------------------------------------------------------------
__global__ void conv_silu_kernel(const float* __restrict__ zx,
                                 const float* __restrict__ conv_w,
                                 const float* __restrict__ conv_b,
                                 float* __restrict__ xbc) {
  int i = blockIdx.x * blockDim.x + threadIdx.x;
  if (i >= ROWS_ * CONVD_) return;
  int c = i % CONVD_;
  int r = i / CONVD_;
  int b = r / L_, l = r % L_;
  float acc = conv_b[c];
#pragma unroll
  for (int k = 0; k < 4; ++k) {
    int ll = l + 3 - k;
    if (ll < L_)
      acc += conv_w[c * 4 + k] * zx[(size_t)(b * L_ + ll) * NPROJ_PAD_ + DI_ + c];
  }
  xbc[(size_t)r * CONVD_ + c] = siluf(acc);
}

// dt = softplus(zx[..., 2176+h] + dt_bias[h])
__global__ void dt_kernel(const float* __restrict__ zx,
                          const float* __restrict__ dt_bias,
                          float* __restrict__ dt) {
  int i = blockIdx.x * blockDim.x + threadIdx.x;
  if (i >= ROWS_ * NH_) return;
  int h = i % NH_, r = i / NH_;
  float v = zx[(size_t)r * NPROJ_PAD_ + DI_ + CONVD_ + h] + dt_bias[h];
  dt[i] = (v > 20.0f) ? v : log1pf(__expf(v));
}

// ---------------------------------------------------------------------------
// Selective scan, reversed time (l = L-1 .. 0). One block per (b,h).
// Thread tid: p = tid>>2 (0..63), owns n in [16*(tid&3), +16) in registers.
// Chunked x16 through LDS, DOUBLE-BUFFERED with GLOBAL_LOAD_ASYNC_TO_LDS
// (ASYNCcnt): chunk c+1 streams into LDS while chunk c's recurrence runs.
// LDS offset = low 32 bits of generic pointer (ISA 10.2: addr[31:0]).
// ---------------------------------------------------------------------------
#define TCH_ 16
#define SCN_C_OFF 1024          // floats
#define SCN_X_OFF 2048
#define SCN_DT_OFF 3072
#define SCN_BUF 3088            // floats per buffer

__global__ void scan_kernel(const float* __restrict__ xbc,
                            const float* __restrict__ dt,
                            const float* __restrict__ A_log,
                            const float* __restrict__ Dp,
                            float* __restrict__ y) {
  __shared__ float smem[2][SCN_BUF];

  const int bh = blockIdx.x;
  const int b = bh >> 4, h = bh & 15;
  const int tid = threadIdx.x;
  const int p = tid >> 2, q = tid & 3, n0 = q * 16;
  const float Aneg = -__expf(A_log[h]);
  const float Dv = Dp[h];
  const int nch = L_ / TCH_;

  float hreg[16];
#pragma unroll
  for (int i = 0; i < 16; ++i) hreg[i] = 0.0f;

  // Issue async loads for chunk cc into buffer buf.
  // 3 tiles (B,C,Xhead) x 16 rows x 64 floats = 768 16-byte transfers,
  // exactly 3 per thread; plus 16 strided b32 loads for dt.
  auto issue_chunk = [&](int cc, int buf) {
#pragma unroll
    for (int t = 0; t < 3; ++t) {
      int idx  = tid + t * 256;
      int tile = idx >> 8;        // 0:B 1:C 2:X
      int rem  = idx & 255;
      int j    = rem >> 4;        // row in chunk
      int ch   = rem & 15;        // 16B chunk in row
      int ll   = L_ - 1 - (cc * TCH_ + j);
      int col  = (tile == 0 ? DI_ : (tile == 1 ? DI_ + DS_ : h * HD_)) + ch * 4;
      const float* g = xbc + (size_t)(b * L_ + ll) * CONVD_ + col;
      unsigned lds = (unsigned)(unsigned long long)(uintptr_t)
                     &smem[buf][tile * 1024 + j * 64 + ch * 4];
      asm volatile("global_load_async_to_lds_b128 %0, %1, off"
                   :: "v"(lds), "v"((unsigned long long)(uintptr_t)g)
                   : "memory");
    }
    if (tid < TCH_) {
      int ll = L_ - 1 - (cc * TCH_ + tid);
      const float* g = dt + (size_t)(b * L_ + ll) * NH_ + h;
      unsigned lds = (unsigned)(unsigned long long)(uintptr_t)
                     &smem[buf][SCN_DT_OFF + tid];
      asm volatile("global_load_async_to_lds_b32 %0, %1, off"
                   :: "v"(lds), "v"((unsigned long long)(uintptr_t)g)
                   : "memory");
    }
  };

  issue_chunk(0, 0);
  asm volatile("s_wait_asynccnt 0x0" ::: "memory");
  __syncthreads();

  for (int cc = 0; cc < nch; ++cc) {
    const int buf = cc & 1;
    if (cc + 1 < nch) issue_chunk(cc + 1, buf ^ 1);

    const float* sB = &smem[buf][0];
    const float* sC = &smem[buf][SCN_C_OFF];
    const float* sX = &smem[buf][SCN_X_OFF];
    const float* sT = &smem[buf][SCN_DT_OFF];

    for (int j = 0; j < TCH_; ++j) {
      float dtv = sT[j];
      float dAv = __expf(dtv * Aneg);
      float xv  = sX[j * 64 + p];
      float dtx = dtv * xv;
      float acc = 0.0f;
#pragma unroll
      for (int i = 0; i < 16; ++i) {
        hreg[i] = hreg[i] * dAv + dtx * sB[j * 64 + n0 + i];
        acc += hreg[i] * sC[j * 64 + n0 + i];
      }
      acc += __shfl_xor(acc, 1, 32);
      acc += __shfl_xor(acc, 2, 32);
      if (q == 0) {
        int ll = L_ - 1 - (cc * TCH_ + j);
        y[(size_t)(b * L_ + ll) * DI_ + h * HD_ + p] = acc + Dv * xv;
      }
    }

    asm volatile("s_wait_asynccnt 0x0" ::: "memory");
    __syncthreads();
  }
}

// ---------------------------------------------------------------------------
// y = y * silu(z); RMS norm over 1024; * rms_w; -> bf16
// ---------------------------------------------------------------------------
__global__ void gated_rms_kernel(const float* __restrict__ y,
                                 const float* __restrict__ zx,
                                 const float* __restrict__ rms_w,
                                 __bf16* __restrict__ ybf) {
  __shared__ float sbuf[256];
  const int r = blockIdx.x, tid = threadIdx.x;
  float g[4];
  float ss = 0.0f;
#pragma unroll
  for (int i = 0; i < 4; ++i) {
    int d = tid + i * 256;
    float zv = zx[(size_t)r * NPROJ_PAD_ + d];
    float gv = y[(size_t)r * DI_ + d] * siluf(zv);
    g[i] = gv;
    ss += gv * gv;
  }
  sbuf[tid] = ss; __syncthreads();
  for (int s = 128; s > 0; s >>= 1) {
    if (tid < s) sbuf[tid] += sbuf[tid + s];
    __syncthreads();
  }
  float scale = rsqrtf(sbuf[0] / (float)DI_ + EPS_);
#pragma unroll
  for (int i = 0; i < 4; ++i) {
    int d = tid + i * 256;
    ybf[(size_t)r * DI_ + d] = (__bf16)(g[i] * scale * rms_w[d]);
  }
}

// LayerNorm over 512 -> bf16
__global__ void layernorm_kernel(const float* __restrict__ x,
                                 const float* __restrict__ gq,
                                 const float* __restrict__ bq,
                                 __bf16* __restrict__ out) {
  __shared__ float sbuf[256];
  const int r = blockIdx.x, tid = threadIdx.x;
  float v0 = x[(size_t)r * DM_ + tid];
  float v1 = x[(size_t)r * DM_ + tid + 256];
  sbuf[tid] = v0 + v1; __syncthreads();
  for (int s = 128; s > 0; s >>= 1) {
    if (tid < s) sbuf[tid] += sbuf[tid + s];
    __syncthreads();
  }
  float m = sbuf[0] / (float)DM_;
  __syncthreads();
  float c0 = v0 - m, c1 = v1 - m;
  sbuf[tid] = c0 * c0 + c1 * c1; __syncthreads();
  for (int s = 128; s > 0; s >>= 1) {
    if (tid < s) sbuf[tid] += sbuf[tid + s];
    __syncthreads();
  }
  float inv = rsqrtf(sbuf[0] / (float)DM_ + EPS_);
  out[(size_t)r * DM_ + tid]       = (__bf16)(c0 * inv * gq[tid] + bq[tid]);
  out[(size_t)r * DM_ + tid + 256] = (__bf16)(c1 * inv * gq[tid + 256] + bq[tid + 256]);
}

// ---------------------------------------------------------------------------
// Host launcher
// ---------------------------------------------------------------------------
extern "C" void kernel_launch(void* const* d_in, const int* in_sizes, int n_in,
                              void* d_out, int out_size, void* d_ws, size_t ws_size,
                              hipStream_t stream) {
  const float* x        = (const float*)d_in[0];
  const float* in_projw = (const float*)d_in[1];
  const float* conv_w   = (const float*)d_in[2];
  const float* conv_b   = (const float*)d_in[3];
  const float* dt_bias  = (const float*)d_in[4];
  const float* A_log    = (const float*)d_in[5];
  const float* Dp       = (const float*)d_in[6];
  const float* rms_w    = (const float*)d_in[7];
  const float* out_projw= (const float*)d_in[8];
  const float* ln_g     = (const float*)d_in[9];
  const float* ln_b     = (const float*)d_in[10];
  const float* w1       = (const float*)d_in[11];
  const float* b1       = (const float*)d_in[12];
  const float* w2       = (const float*)d_in[13];
  const float* b2       = (const float*)d_in[14];

  char* ws = (char*)d_ws;
  constexpr size_t OFF_XBF  = 0;                          // 16384*512   bf16
  constexpr size_t OFF_INW  = OFF_XBF  + 16777216ull;     // 2240*512    bf16
  constexpr size_t OFF_ZX   = OFF_INW  + 2293760ull;      // 16384*2240  f32
  constexpr size_t OFF_XBC  = OFF_ZX   + 146800640ull;    // 16384*1152  f32
  constexpr size_t OFF_DT   = OFF_XBC  + 75497472ull;     // 16384*16    f32
  constexpr size_t OFF_Y    = OFF_DT   + 1048576ull;      // 16384*1024  f32
  constexpr size_t OFF_YBF  = OFF_Y    + 67108864ull;     // 16384*1024  bf16
  constexpr size_t OFF_OPW  = OFF_YBF  + 33554432ull;     // 512*1024    bf16
  constexpr size_t OFF_T1   = OFF_OPW  + 1048576ull;      // 16384*512   f32
  constexpr size_t OFF_TN   = OFF_T1   + 33554432ull;     // 16384*512   bf16
  constexpr size_t OFF_W1C  = OFF_TN   + 16777216ull;     // 1024*512    bf16
  constexpr size_t OFF_HMID = OFF_W1C  + 1048576ull;      // 16384*1024  bf16
  constexpr size_t OFF_W2B  = OFF_HMID + 33554432ull;     // 512*1024    bf16

  __bf16* xbf  = (__bf16*)(ws + OFF_XBF);
  __bf16* inw  = (__bf16*)(ws + OFF_INW);
  float*  zx   = (float*)(ws + OFF_ZX);
  float*  xbc  = (float*)(ws + OFF_XBC);
  float*  dt   = (float*)(ws + OFF_DT);
  float*  y    = (float*)(ws + OFF_Y);
  __bf16* ybf  = (__bf16*)(ws + OFF_YBF);
  __bf16* opw  = (__bf16*)(ws + OFF_OPW);
  float*  t1   = (float*)(ws + OFF_T1);
  __bf16* tn   = (__bf16*)(ws + OFF_TN);
  __bf16* w1c  = (__bf16*)(ws + OFF_W1C);
  __bf16* hmid = (__bf16*)(ws + OFF_HMID);
  __bf16* w2b  = (__bf16*)(ws + OFF_W2B);

  const int TPB = 256;

  // --- weight / input prep (bf16) ---
  cvt_f32_bf16_kernel<<<(ROWS_ * DM_ + TPB - 1) / TPB, TPB, 0, stream>>>(x, xbf, ROWS_ * DM_);
  pad_inproj_kernel<<<(NPROJ_PAD_ * DM_ + TPB - 1) / TPB, TPB, 0, stream>>>(in_projw, inw);
  cvt_f32_bf16_kernel<<<(DM_ * DI_ + TPB - 1) / TPB, TPB, 0, stream>>>(out_projw, opw, DM_ * DI_);
  fold_w1_kernel<<<(1024 * DM_ + TPB - 1) / TPB, TPB, 0, stream>>>(w1, w1c);
  cvt_f32_bf16_kernel<<<(DM_ * DI_ + TPB - 1) / TPB, TPB, 0, stream>>>(w2, w2b, DM_ * DI_);

  // --- GEMM1: zxbcdt = x @ in_proj_w^T  (16384 x 2240, padded) ---
  gemm_bf16_wmma_kernel<<<dim3(ROWS_ / 128, NPROJ_PAD_ / 64), TPB, 0, stream>>>(
      xbf, inw, nullptr, zx, nullptr, DM_, NPROJ_PAD_, 0);

  // --- conv + silu, dt = softplus ---
  conv_silu_kernel<<<(ROWS_ * CONVD_ + TPB - 1) / TPB, TPB, 0, stream>>>(zx, conv_w, conv_b, xbc);
  dt_kernel<<<(ROWS_ * NH_ + TPB - 1) / TPB, TPB, 0, stream>>>(zx, dt_bias, dt);

  // --- selective scan (reversed time, async double-buffered LDS) ---
  scan_kernel<<<B_ * NH_, TPB, 0, stream>>>(xbc, dt, A_log, Dp, y);

  // --- gated RMS norm -> bf16 ---
  gated_rms_kernel<<<ROWS_, TPB, 0, stream>>>(y, zx, rms_w, ybf);

  // --- GEMM2: t1 = ybf @ out_proj_w^T  (16384 x 512) ---
  gemm_bf16_wmma_kernel<<<dim3(ROWS_ / 128, DM_ / 64), TPB, 0, stream>>>(
      ybf, opw, nullptr, t1, nullptr, DI_, DM_, 0);

  // --- layernorm -> bf16 ---
  layernorm_kernel<<<ROWS_, TPB, 0, stream>>>(t1, ln_g, ln_b, tn);

  // --- GEMM3 (cat folded): hmid = silu(tn @ w1c^T + b1) -> bf16 ---
  gemm_bf16_wmma_kernel<<<dim3(ROWS_ / 128, 1024 / 64), TPB, 0, stream>>>(
      tn, w1c, b1, nullptr, hmid, DM_, 1024, 1);

  // --- GEMM4: out = hmid @ w2^T + b2 -> f32 ---
  gemm_bf16_wmma_kernel<<<dim3(ROWS_ / 128, DM_ / 64), TPB, 0, stream>>>(
      hmid, w2b, b2, (float*)d_out, nullptr, DI_, DM_, 0);
}
